// NetworkStitch_5016521802529
// MI455X (gfx1250) — matrix operand order repaired and checked by reading, not compiled, not run on hardware
//
#include <hip/hip_runtime.h>
#include <hip/hip_bf16.h>

typedef _Float16 h16;
typedef __attribute__((ext_vector_type(8)))  _Float16 v8h;
typedef __attribute__((ext_vector_type(16))) _Float16 v16h;
typedef __attribute__((ext_vector_type(8)))  float    v8f;

#define CCH 64      // channels
#define NPX 64      // H == W == 64
#define SR  4       // search range
#define ND  9       // 2*SR+1 offsets per direction

// One wave32 per (batch, line, direction).
//   mode 0 (horizontal): inner axis = w (stride 1),  outer = h (stride 64), out chans 0..8
//   mode 1 (vertical):   inner axis = h (stride 64), outer = w (stride 1),  out chans 9..17
// Correlation tile via WMMA: D[16x32] = A[16px x 64ch] * B[64ch x 32 shifted px],
// then cost[m,d] = D[m][m+d] * invnorm[m], leaky-relu.
__global__ __launch_bounds__(32)
void costvol_wmma_kernel(const float* __restrict__ f1,
                         const float* __restrict__ f2,
                         float* __restrict__ out)
{
    __shared__ h16   f1T[NPX][72];           // [inner pixel][channel], f16, 144B rows (16B aligned chunks)
    __shared__ h16   f2T[NPX + 2 * SR][72];  // padded along inner axis
    __shared__ float invA[NPX];              // 1 / (64 * max(||f1(px)||, eps))
    __shared__ float Dbuf[16][33];           // WMMA result scatter (padded to dodge bank conflicts)

    const int lane  = threadIdx.x;           // 0..31
    int x = blockIdx.x;
    const int mode  = x & 1;  x >>= 1;       // hori/verti interleaved so same image stays hot in L2
    const int outer = x & 63;
    const int b     = x >> 6;

    const int sIn   = mode ? NPX : 1;
    const int sOut  = mode ? 1 : NPX;
    const int cbase = mode ? ND : 0;

    const int ibase = b * (CCH * NPX * NPX) + outer * sOut;

    // ---------------- stage both feature lines into LDS (f16), sumsq of f1 in fp32 ----------------
    const int i0 = lane, i1 = lane + 32;     // each lane owns 2 inner pixels
    float ss0 = 0.f, ss1 = 0.f;
    for (int c = 0; c < CCH; ++c) {
        const int o = ibase + c * (NPX * NPX);
        float a0 = f1[o + i0 * sIn];
        float a1 = f1[o + i1 * sIn];
        ss0 += a0 * a0;
        ss1 += a1 * a1;
        f1T[i0][c] = (h16)a0;
        f1T[i1][c] = (h16)a1;
        float b0 = f2[o + i0 * sIn];
        float b1 = f2[o + i1 * sIn];
        f2T[SR + i0][c] = (h16)b0;
        f2T[SR + i1][c] = (h16)b1;
    }
    // zero-pad borders of f2 line (pixels -SR..-1 and NPX..NPX+SR-1)
    for (int r = 0; r < 2 * SR; ++r) {
        const int row = (r < SR) ? r : (NPX + r);   // 0..3 and 68..71
        for (int c = lane; c < CCH; c += 32) f2T[row][c] = (h16)0.f;
    }
    {
        float n0 = fmaxf(sqrtf(ss0), 1e-12f);
        float n1 = fmaxf(sqrtf(ss1), 1e-12f);
        invA[i0] = 1.0f / (64.0f * n0);
        invA[i1] = 1.0f / (64.0f * n1);
    }
    __syncthreads();

    // ---------------- 4 tiles of 16 inner pixels ----------------
    const int m       = lane & 15;
    const bool hiHalf = lane >= 16;

    for (int t = 0; t < 4; ++t) {
        const int w0 = t * 16;

        // A fragments (16x32 f16, MxK): lanes 0-15 row m hold K {kb..kb+7, kb+16..kb+23}, kb=0;
        // lanes 16-31 kb=8.  f1T rows are pixel-major so chunks are contiguous 16B.
        v16h A0, A1;
        {
            const v8h* rowp = (const v8h*)f1T[w0 + m];
            const int c0 = hiHalf ? 1 : 0;
            v8h p0 = rowp[c0];       // K kb..kb+7        (WMMA #1)
            v8h p1 = rowp[c0 + 2];   // K kb+16..kb+23
            v8h p2 = rowp[c0 + 4];   // K kb+32..kb+39    (WMMA #2, +32)
            v8h p3 = rowp[c0 + 6];   // K kb+48..kb+55
#pragma unroll
            for (int i = 0; i < 8; ++i) {
                A0[i] = p0[i]; A0[i + 8] = p1[i];
                A1[i] = p2[i]; A1[i + 8] = p3[i];
            }
        }

        // B fragments (32x16 f16, KxN): lane column j=m; lanes 0-15 K {0..7,16..23}, lanes 16-31 K {8..15,24..31}.
        // Left tile covers shifted pixels w0-4..w0+11, right tile w0+12..w0+19 (+don't-cares).
        v16h BL0, BL1, BR0, BR1;
        {
            const int pL = w0 + m;                              // padded index, always in range
            int       pR = w0 + 16 + m;
            if (pR > NPX + 2 * SR - 1) pR = NPX + 2 * SR - 1;   // clamp; clamped cols never extracted
            const v8h* rl = (const v8h*)f2T[pL];
            const v8h* rr = (const v8h*)f2T[pR];
            const int kc = hiHalf ? 1 : 0;
            v8h l0 = rl[kc], l1 = rl[kc + 2], l4 = rl[kc + 4], l5 = rl[kc + 6];
            v8h r0 = rr[kc], r1 = rr[kc + 2], r4 = rr[kc + 4], r5 = rr[kc + 6];
#pragma unroll
            for (int i = 0; i < 8; ++i) {
                BL0[i] = l0[i]; BL0[i + 8] = l1[i];
                BL1[i] = l4[i]; BL1[i + 8] = l5[i];
                BR0[i] = r0[i]; BR0[i + 8] = r1[i];
                BR1[i] = r4[i]; BR1[i + 8] = r5[i];
            }
        }

        // D = A(16x64) x B(64x32) via 4 chained f16 WMMAs with fp32 accumulate
        v8f accL = {};
        v8f accR = {};
        accL = __builtin_amdgcn_wmma_f32_16x16x32_f16(false, A0, false, BL0, (short)0, accL, false, false);
        accL = __builtin_amdgcn_wmma_f32_16x16x32_f16(false, A1, false, BL1, (short)0, accL, false, false);
        accR = __builtin_amdgcn_wmma_f32_16x16x32_f16(false, A0, false, BR0, (short)0, accR, false, false);
        accR = __builtin_amdgcn_wmma_f32_16x16x32_f16(false, A1, false, BR1, (short)0, accR, false, false);

        // scatter D to LDS: C/D layout -> lane n=m holds rows {mb..mb+7} in VGPRs 0..7
        {
            const int mb = hiHalf ? 8 : 0;
#pragma unroll
            for (int r = 0; r < 8; ++r) {
                Dbuf[mb + r][m]      = accL[r];
                Dbuf[mb + r][16 + m] = accR[r];
            }
        }
        __syncthreads();

        // extract 16 pixels x 9 offsets: cost[mm,d] = D[mm][mm+d] * invnorm, leaky-relu 0.01
        for (int idx = lane; idx < 16 * ND; idx += 32) {
            const int mm = idx & 15;
            const int d  = idx >> 4;
            float v = Dbuf[mm][mm + d] * invA[w0 + mm];
            v = (v >= 0.f) ? v : 0.01f * v;
            const int oo = b * (2 * ND * NPX * NPX)
                         + (cbase + d) * (NPX * NPX)
                         + (w0 + mm) * sIn + outer * sOut;
            out[oo] = v;
        }
        __syncthreads();
    }
}

extern "C" void kernel_launch(void* const* d_in, const int* in_sizes, int n_in,
                              void* d_out, int out_size, void* d_ws, size_t ws_size,
                              hipStream_t stream) {
    (void)in_sizes; (void)n_in; (void)out_size; (void)d_ws; (void)ws_size;
    const float* f1 = (const float*)d_in[0];
    const float* f2 = (const float*)d_in[1];
    float* out = (float*)d_out;
    // 256 batches * 64 lines * 2 directions, one wave32 per block;
    // hori/verti of the same image adjacent so direction #2 hits L2.
    dim3 grid(256 * 64 * 2), block(32);
    costvol_wmma_kernel<<<grid, block, 0, stream>>>(f1, f2, out);
}
